// GNNEncoder_85392539779175
// MI455X (gfx1250) — matrix operand order repaired and checked by reading, not compiled, not run on hardware
//
#include <hip/hip_runtime.h>
#include <hip/hip_bf16.h>

typedef __attribute__((ext_vector_type(16))) __bf16 v16bf;
typedef __attribute__((ext_vector_type(8)))  __bf16 v8bf;
typedef __attribute__((ext_vector_type(8)))  float  v8f;

#define WAVES_PER_BLOCK 8
#define GEMM_THREADS 256

// ---------------------------------------------------------------------------
// Degree / normalization kernels
// ---------------------------------------------------------------------------
__global__ void deg_zero_kernel(int* __restrict__ deg, int n) {
    int i = blockIdx.x * blockDim.x + threadIdx.x;
    if (i < n) deg[i] = 0;
}

__global__ void deg_count_kernel(const int* __restrict__ dst, int E,
                                 int* __restrict__ deg) {
    int i = blockIdx.x * blockDim.x + threadIdx.x;
    if (i < E) atomicAdd(&deg[dst[i]], 1);
}

__global__ void dinv_kernel(const int* __restrict__ deg,
                            float* __restrict__ dinv, int n) {
    int i = blockIdx.x * blockDim.x + threadIdx.x;
    if (i < n) {
        // self-loop adds 1 -> degree always > 0
        dinv[i] = rsqrtf((float)deg[i] + 1.0f);
    }
}

// ---------------------------------------------------------------------------
// GEMM: H[M,N] = (relu? max(X,0):X)[M,K] @ W[K,N]   via v_wmma_f32_16x16x32_bf16
// Block = 16 rows of X, 8 waves. nTiles = N/16 is 8 or 16 -> 1 or 2 tiles/wave.
// M must be a multiple of 16 (20000 = 1250*16), K a multiple of 32.
// ---------------------------------------------------------------------------
__global__ __launch_bounds__(GEMM_THREADS)
void gemm_bf16_kernel(const float* __restrict__ X, const float* __restrict__ W,
                      float* __restrict__ H, int M, int K, int N, int relu) {
    __shared__ __bf16 aT[16 * 32];    // A tile, row-major [16][32]
    __shared__ __bf16 wT[256 * 32];   // W slab transposed: [n][kk], stride 32

    const int tid  = threadIdx.x;
    const int lane = tid & 31;
    const int wave = tid >> 5;
    const int row0 = blockIdx.x * 16;
    const int nTiles = N >> 4;        // 8 (N=128) or 16 (N=256)

    v8f acc0 = {0.f,0.f,0.f,0.f,0.f,0.f,0.f,0.f};
    v8f acc1 = {0.f,0.f,0.f,0.f,0.f,0.f,0.f,0.f};

    for (int kb = 0; kb < K; kb += 32) {
        // ---- stage A tile (16x32 fp32 -> bf16), ReLU fused on input ----
        for (int i = tid; i < 16 * 32; i += GEMM_THREADS) {
            int r = i >> 5, c = i & 31;
            float v = X[(row0 + r) * K + kb + c];
            if (relu) v = fmaxf(v, 0.0f);
            aT[i] = (__bf16)v;
        }
        // ---- stage W slab transposed: wT[n*32 + kk] = W[(kb+kk)*N + n] ----
        for (int i = tid; i < 32 * N; i += GEMM_THREADS) {
            int kk = i / N;
            int n  = i - kk * N;
            wT[n * 32 + kk] = (__bf16)W[(kb + kk) * N + n];
        }
        __syncthreads();

        // ---- A fragment (16x32, contiguous K per lane half, same order as B) ----
        // lanes 0-15: M=lane, K = 0..15 (elts 0-15)
        // lanes 16-31: M=lane-16, K = 16..31
        const int r  = lane & 15;
        const int kg = (lane >> 4) * 16;
        v8bf alo = *(const v8bf*)&aT[r * 32 + kg];
        v8bf ahi = *(const v8bf*)&aT[r * 32 + kg + 8];
        v16bf a = __builtin_shufflevector(alo, ahi,
                    0,1,2,3,4,5,6,7,8,9,10,11,12,13,14,15);

        // ---- B fragments (col = lane&15, K half selected by lane>=16) ----
        const int kh = (lane >> 4) * 16;
        {
            int n = wave * 16 + (lane & 15);
            v16bf b = *(const v16bf*)&wT[n * 32 + kh];
            acc0 = __builtin_amdgcn_wmma_f32_16x16x32_bf16(
                       false, a, false, b, (short)0, acc0, false, false);
        }
        if (nTiles > 8) {
            int n = (wave + 8) * 16 + (lane & 15);
            v16bf b = *(const v16bf*)&wT[n * 32 + kh];
            acc1 = __builtin_amdgcn_wmma_f32_16x16x32_bf16(
                       false, a, false, b, (short)0, acc1, false, false);
        }
        __syncthreads();
    }

    // ---- store D (lane 0-15: rows v+0, lanes 16-31: rows v+8, col=lane&15) --
    const int rbase = row0 + ((lane >> 4) << 3);
    {
        int col = wave * 16 + (lane & 15);
        #pragma unroll
        for (int v = 0; v < 8; ++v)
            H[(rbase + v) * N + col] = acc0[v];
    }
    if (nTiles > 8) {
        int col = (wave + 8) * 16 + (lane & 15);
        #pragma unroll
        for (int v = 0; v < 8; ++v)
            H[(rbase + v) * N + col] = acc1[v];
    }
}

// ---------------------------------------------------------------------------
// Aggregation: out[i,j] = h[i,j]*dinv[i]^2 + b[j]  (self loop + bias, full init)
// ---------------------------------------------------------------------------
__global__ void agg_init_kernel(const float* __restrict__ h,
                                const float* __restrict__ dinv,
                                const float* __restrict__ bias,
                                float* __restrict__ out,
                                int M, int N) {
    unsigned gid = blockIdx.x * blockDim.x + threadIdx.x;
    unsigned total = (unsigned)M * (unsigned)N;
    if (gid < total) {
        unsigned r = gid / (unsigned)N;
        unsigned c = gid - r * (unsigned)N;
        float di = dinv[r];
        out[gid] = h[gid] * di * di + bias[c];
    }
}

// out[dst] += h[src] * dinv[src] * dinv[dst] ; 1 thread = 1 edge x 4 features
__global__ void agg_scatter_kernel(const int* __restrict__ src,
                                   const int* __restrict__ dst,
                                   const float* __restrict__ dinv,
                                   const float* __restrict__ h,
                                   float* __restrict__ out,
                                   int E, int N) {
    unsigned chunks = (unsigned)(N >> 2);        // float4 chunks per row
    unsigned gid = blockIdx.x * blockDim.x + threadIdx.x;
    unsigned total = (unsigned)E * chunks;
    if (gid >= total) return;
    unsigned e = gid / chunks;
    unsigned f = (gid - e * chunks) << 2;
    int s = src[e];
    int d = dst[e];
    float nm = dinv[s] * dinv[d];
    const float4 hv = *(const float4*)&h[(unsigned)s * (unsigned)N + f];
    float* o = &out[(unsigned)d * (unsigned)N + f];
    unsafeAtomicAdd(o + 0, hv.x * nm);
    unsafeAtomicAdd(o + 1, hv.y * nm);
    unsafeAtomicAdd(o + 2, hv.z * nm);
    unsafeAtomicAdd(o + 3, hv.w * nm);
}

// ---------------------------------------------------------------------------
// Host launch
// ---------------------------------------------------------------------------
extern "C" void kernel_launch(void* const* d_in, const int* in_sizes, int n_in,
                              void* d_out, int out_size, void* d_ws, size_t ws_size,
                              hipStream_t stream) {
    const int* ei  = (const int*)d_in[0];      // [2, E] int32
    const float* x0 = (const float*)d_in[1];   // [M, 128]
    const float* W1 = (const float*)d_in[2];   // [128, 256]
    const float* b1 = (const float*)d_in[3];
    const float* W2 = (const float*)d_in[4];   // [256, 256]
    const float* b2 = (const float*)d_in[5];
    const float* W3 = (const float*)d_in[6];   // [256, 128]
    const float* b3 = (const float*)d_in[7];
    float* out = (float*)d_out;                // [M, 128]

    const int E = in_sizes[0] / 2;             // 640000
    const int M = in_sizes[1] / 128;           // 20000
    const int* src = ei;
    const int* dst = ei + E;

    // Workspace layout (bytes): dinv | deg | bufH | bufA  (~41 MB total)
    char* ws = (char*)d_ws;
    float* dinv = (float*)(ws + 0);                       // M floats
    int*   deg  = (int*)  (ws + 131072);                  // M ints
    float* bufH = (float*)(ws + 262144);                  // M*256 floats
    float* bufA = (float*)(ws + 262144 + (size_t)M * 256 * sizeof(float));

    const int T = 256;
    // --- normalization (shared by all 3 layers) ---
    deg_zero_kernel <<<(M + T - 1) / T, T, 0, stream>>>(deg, M);
    deg_count_kernel<<<(E + T - 1) / T, T, 0, stream>>>(dst, E, deg);
    dinv_kernel     <<<(M + T - 1) / T, T, 0, stream>>>(deg, dinv, M);

    const int gemmBlocks = M / 16;             // 1250
    // --- layer 1: x0[ M,128] @ W1 -> bufH[M,256]; aggregate -> bufA ---
    gemm_bf16_kernel<<<gemmBlocks, GEMM_THREADS, 0, stream>>>(x0, W1, bufH, M, 128, 256, 0);
    agg_init_kernel  <<<((unsigned)M * 256 + T - 1) / T, T, 0, stream>>>(bufH, dinv, b1, bufA, M, 256);
    agg_scatter_kernel<<<((unsigned)E * 64 + T - 1) / T, T, 0, stream>>>(src, dst, dinv, bufH, bufA, E, 256);

    // --- layer 2: relu(bufA) @ W2 -> bufH[M,256]; aggregate -> bufA ---
    gemm_bf16_kernel<<<gemmBlocks, GEMM_THREADS, 0, stream>>>(bufA, W2, bufH, M, 256, 256, 1);
    agg_init_kernel  <<<((unsigned)M * 256 + T - 1) / T, T, 0, stream>>>(bufH, dinv, b2, bufA, M, 256);
    agg_scatter_kernel<<<((unsigned)E * 64 + T - 1) / T, T, 0, stream>>>(src, dst, dinv, bufH, bufA, E, 256);

    // --- layer 3: relu(bufA) @ W3 -> bufH[M,128]; aggregate -> d_out ---
    gemm_bf16_kernel<<<gemmBlocks, GEMM_THREADS, 0, stream>>>(bufA, W3, bufH, M, 256, 128, 1);
    agg_init_kernel  <<<((unsigned)M * 128 + T - 1) / T, T, 0, stream>>>(bufH, dinv, b3, out, M, 128);
    agg_scatter_kernel<<<((unsigned)E * 32 + T - 1) / T, T, 0, stream>>>(src, dst, dinv, bufH, out, E, 128);
}
